// EAA_73933567033420
// MI455X (gfx1250) — matrix operand order repaired and checked by reading, not compiled, not run on hardware
//
#include <hip/hip_runtime.h>
#include <hip/hip_bf16.h>

typedef __attribute__((ext_vector_type(16))) _Float16 v16h;
typedef __attribute__((ext_vector_type(8)))  _Float16 v8h;
typedef __attribute__((ext_vector_type(8)))  float    v8f;
typedef __attribute__((ext_vector_type(4)))  float    v4f;

constexpr int kN  = 8000;     // nodes
constexpr int kE  = 64000;    // edges
constexpr int kH  = 128;      // hidden
constexpr int kF  = 128;      // filters
constexpr int kHD = 10;       // heads
constexpr int kG  = 50;       // gaussians (padded to 64)
constexpr int kGP = 64;       // padded gaussian dim
constexpr int kL  = 2;        // layers
constexpr int kNG = 64;       // graphs
constexpr int kHF = kHD * kF; // 1280

// shifted softplus via raw hw transcendentals (no denorm fixups needed:
// log argument is in [1,2]):
//   ssp(x) = max(x,0) + (log2(1 + 2^(-|x|*log2e)) - 1) * ln2
__device__ __forceinline__ float ssp_f(float x) {
  float t  = __builtin_amdgcn_exp2f(-fabsf(x) * 1.44269504088896341f);
  float lg = __builtin_amdgcn_logf(1.0f + t);
  return fmaxf(x, 0.0f) + (lg - 1.0f) * 0.69314718055994531f;
}

// ---------------------------------------------------------------------------
// A-fragment (16x32 f16) from f32 row-major A, row = lane&15, via float4 loads.
// ---------------------------------------------------------------------------
__device__ __forceinline__ v16h load_afrag_f32(const float* __restrict__ Arow,
                                               int kc, int hi) {
  v16h a;
  const v4f x0 = *(const v4f*)(Arow + kc + hi * 8);
  const v4f x1 = *(const v4f*)(Arow + kc + hi * 8 + 4);
  const v4f x2 = *(const v4f*)(Arow + kc + hi * 8 + 16);
  const v4f x3 = *(const v4f*)(Arow + kc + hi * 8 + 20);
#pragma unroll
  for (int i = 0; i < 4; ++i) {
    a[i]      = (_Float16)x0[i];
    a[4 + i]  = (_Float16)x1[i];
    a[8 + i]  = (_Float16)x2[i];
    a[12 + i] = (_Float16)x3[i];
  }
  return a;
}

// ---------------------------------------------------------------------------
// Generic GEMM: C[M,N] = act(A[M,K] @ B[K,N] + bias) (+ resid)
// B supplied pre-transposed+f16: BT[N,K]. One wave computes a 16x64 strip
// (4 WMMA tiles sharing one A fragment). K compile-time, multiple of 32.
// ---------------------------------------------------------------------------
template <int K, int ACT, bool BIAS, bool RESID>
__global__ __launch_bounds__(256)
void gat_gemm_wmma(const float* __restrict__ A, const _Float16* __restrict__ BT,
                   const float* __restrict__ bias, const float* __restrict__ resid,
                   float* __restrict__ C, int M, int N) {
  int tid  = threadIdx.x;
  int wave = (blockIdx.x * blockDim.x + tid) >> 5;
  int lane = tid & 31;
  int strips_n = N >> 6;                       // 64-wide strips
  int strips   = (M >> 4) * strips_n;
  if (wave >= strips) return;                  // wave-uniform
  int tm = wave / strips_n, tn4 = wave % strips_n;
  int hi = lane >> 4, lo = lane & 15;

  const float* Arow = A + (size_t)(tm * 16 + lo) * K;
  const _Float16* Bbase[4];
#pragma unroll
  for (int j = 0; j < 4; ++j)
    Bbase[j] = BT + (size_t)(tn4 * 64 + j * 16 + lo) * K;

  v8f c[4] = {};
  for (int kc = 0; kc < K; kc += 32) {
    v16h a = load_afrag_f32(Arow, kc, hi);
#pragma unroll
    for (int j = 0; j < 4; ++j) {
      v8h b0 = *(const v8h*)(Bbase[j] + kc + hi * 8);  // k = 8h+s, 16+8h+s
      v8h b1 = *(const v8h*)(Bbase[j] + kc + hi * 8 + 16);
      v16h b;
#pragma unroll
      for (int i = 0; i < 8; ++i) { b[i] = b0[i]; b[8 + i] = b1[i]; }
      c[j] = __builtin_amdgcn_wmma_f32_16x16x32_f16(false, a, false, b,
                                                    (short)0, c[j], false, false);
    }
  }

#pragma unroll
  for (int j = 0; j < 4; ++j) {
    int n = tn4 * 64 + j * 16 + lo;
#pragma unroll
    for (int v = 0; v < 8; ++v) {
      int m = tm * 16 + v + 8 * hi;
      float val = c[j][v];
      if (BIAS)  val += bias[n];
      if (ACT)   val  = ssp_f(val);
      if (RESID) val += resid[(size_t)m * N + n];
      C[(size_t)m * N + n] = val;
    }
  }
}

// ---------------------------------------------------------------------------
// Fused per-edge pipeline for a 16-edge tile (8 waves / block):
//   hidden = ssp(demb16 @ w0 + b0)   -> LDS f16 [16][128]
//   per head: W_h = hidden @ w2 + b2 (registers), eedge reduce (ds_add_f32),
//             a = ssp(left+eedge+right), scatter vl[row]*a*W_h into agg.
// A-fragments for stage 2 are hoisted to registers once (head-invariant).
// ---------------------------------------------------------------------------
__global__ __launch_bounds__(256)
void gat_edge_fused(const _Float16* __restrict__ demb16, const int* __restrict__ row,
                    const int* __restrict__ col,
                    const _Float16* __restrict__ w0T,   // [128, 64]
                    const float* __restrict__ b0,
                    const _Float16* __restrict__ w2T,   // [1280, 128]
                    const float* __restrict__ b2,
                    const float* __restrict__ ae,
                    const float* __restrict__ leftb, const float* __restrict__ rightb,
                    const float* __restrict__ vl, float* __restrict__ agg) {
  __shared__ _Float16 hid[16][kH];
  __shared__ float eedge[16];
  __shared__ float ash[16];
  __shared__ int   rj[16];     // row (source j) indices for this tile
  __shared__ int   ci[16];     // col (dest i) indices for this tile

  int tid = threadIdx.x;
  int w = tid >> 5, lane = tid & 31;
  int hi = lane >> 4, lo = lane & 15;
  int e0 = blockIdx.x * 16;

  if (tid < 16) { rj[tid] = row[e0 + tid]; ci[tid] = col[e0 + tid]; }

  // ---- stage 1: hidden (wave w owns columns [w*16, w*16+16)) ----
  {
    const _Float16* Arow = demb16 + (size_t)(e0 + lo) * kGP;
    int n = w * 16 + lo;
    const _Float16* Brow = w0T + (size_t)n * kGP;
    v8f c = {};
#pragma unroll
    for (int kc = 0; kc < kGP; kc += 32) {
      v16h a, b;
      v8h a0 = *(const v8h*)(Arow + kc + hi * 8);
      v8h a1 = *(const v8h*)(Arow + kc + hi * 8 + 16);
#pragma unroll
      for (int i = 0; i < 8; ++i) { a[i] = a0[i]; a[8 + i] = a1[i]; }
      v8h b0 = *(const v8h*)(Brow + kc + hi * 8);
      v8h b1 = *(const v8h*)(Brow + kc + hi * 8 + 16);
#pragma unroll
      for (int i = 0; i < 8; ++i) { b[i] = b0[i]; b[8 + i] = b1[i]; }
      c = __builtin_amdgcn_wmma_f32_16x16x32_f16(false, a, false, b,
                                                 (short)0, c, false, false);
    }
#pragma unroll
    for (int v = 0; v < 8; ++v) {
      int m = v + 8 * hi;
      hid[m][n] = (_Float16)ssp_f(c[v] + b0[n]);
    }
  }
  __syncthreads();

  // ---- hoist stage-2 A fragments (head-invariant) into registers ----
  v16h afr[kH / 32];
#pragma unroll
  for (int kc = 0; kc < kH; kc += 32) {
    v8h a0 = *(const v8h*)(&hid[lo][kc + hi * 8]);          // ds_load_b128
    v8h a1 = *(const v8h*)(&hid[lo][kc + hi * 8 + 16]);
#pragma unroll
    for (int i = 0; i < 8; ++i) {
      afr[kc / 32][i] = a0[i];
      afr[kc / 32][8 + i] = a1[i];
    }
  }

  // ---- stage 2: per-head filter, attention, scatter ----
  int nf = w * 16 + lo;                 // f within head
  for (int h = 0; h < kHD; ++h) {
    if (tid < 16) eedge[tid] = 0.0f;
    __syncthreads();

    int n = h * kF + nf;                // column in [0,1280)
    const _Float16* Brow = w2T + (size_t)n * kH;
    v8f c = {};
#pragma unroll
    for (int kc = 0; kc < kH; kc += 32) {
      v8h b0 = *(const v8h*)(Brow + kc + hi * 8);
      v8h b1 = *(const v8h*)(Brow + kc + hi * 8 + 16);
      v16h b;
#pragma unroll
      for (int i = 0; i < 8; ++i) { b[i] = b0[i]; b[8 + i] = b1[i]; }
      c = __builtin_amdgcn_wmma_f32_16x16x32_f16(false, afr[kc / 32], false, b,
                                                 (short)0, c, false, false);
    }

    float aev = ae[h * kF + nf];
    float bias2 = b2[n];
#pragma unroll
    for (int v = 0; v < 8; ++v) {
      c[v] += bias2;
      atomicAdd(&eedge[v + 8 * hi], c[v] * aev);    // ds_add_f32 reduce
    }
    __syncthreads();

    if (tid < 16) {
      ash[tid] = ssp_f(leftb[rj[tid] * kHD + h] + eedge[tid] +
                       rightb[ci[tid] * kHD + h]);
    }
    __syncthreads();

#pragma unroll
    for (int v = 0; v < 8; ++v) {
      int m = v + 8 * hi;
      float e = vl[(size_t)rj[m] * kHF + n] * ash[m] * c[v];
      atomicAdd(&agg[(size_t)ci[m] * kHF + n], e);  // L2-resident scatter
    }
    __syncthreads();
  }
}

// ---------------------------------------------------------------------------
// Weight transpose + f16 convert: BT[n*Kpad + k] = (k<Kin) ? B[k*N+n] : 0
// ---------------------------------------------------------------------------
__global__ void gat_conv_bt16(const float* __restrict__ B, _Float16* __restrict__ BT,
                              int Kin, int N, int Kpad) {
  int idx = blockIdx.x * blockDim.x + threadIdx.x;
  if (idx >= N * Kpad) return;
  int n = idx / Kpad, k = idx % Kpad;
  BT[idx] = (k < Kin) ? (_Float16)B[(size_t)k * N + n] : (_Float16)0.0f;
}

// ---------------------------------------------------------------------------
// Gaussian smearing -> f16, zero-padded to [E, 64]
// ---------------------------------------------------------------------------
__global__ void gat_gsmear(const float* __restrict__ pos, const int* __restrict__ row,
                           const int* __restrict__ col, _Float16* __restrict__ demb16) {
  int e = blockIdx.x * blockDim.x + threadIdx.x;
  if (e >= kE) return;
  int j = row[e], i = col[e];
  float dx = pos[j * 3 + 0] - pos[i * 3 + 0];
  float dy = pos[j * 3 + 1] - pos[i * 3 + 1];
  float dz = pos[j * 3 + 2] - pos[i * 3 + 2];
  float d = sqrtf(dx * dx + dy * dy + dz * dz);
  const float step = 10.0f / 49.0f;
  const float coeff = -0.5f / (step * step) * 1.44269504088896341f; // pre-scale for exp2
#pragma unroll 8
  for (int g = 0; g < kGP; ++g) {
    float t = d - step * (float)g;
    float val = (g < kG) ? __builtin_amdgcn_exp2f(coeff * t * t) : 0.0f;
    demb16[(size_t)e * kGP + g] = (_Float16)val;
  }
}

__global__ void gat_embed(const int* __restrict__ z, const float* __restrict__ table,
                          float* __restrict__ v) {
  int idx = blockIdx.x * blockDim.x + threadIdx.x;
  if (idx >= kN * kH) return;
  int n = idx / kH, c = idx % kH;
  v[idx] = table[z[n] * kH + c];
}

__global__ void gat_leftright(const float* __restrict__ vl, const float* __restrict__ al,
                              const float* __restrict__ ar, float* __restrict__ leftb,
                              float* __restrict__ rightb) {
  int idx = blockIdx.x * blockDim.x + threadIdx.x;
  if (idx >= kN * kHD) return;
  int n = idx / kHD, h = idx % kHD;
  const float* vp  = vl + (size_t)n * kHF + h * kF;
  const float* alp = al + h * kF;
  const float* arp = ar + h * kF;
  float sl = 0.0f, sr = 0.0f;
#pragma unroll 4
  for (int f = 0; f < kF; f += 4) {
    v4f x = *(const v4f*)(vp + f);
    v4f l = *(const v4f*)(alp + f);
    v4f r = *(const v4f*)(arp + f);
    sl += x[0]*l[0] + x[1]*l[1] + x[2]*l[2] + x[3]*l[3];
    sr += x[0]*r[0] + x[1]*r[1] + x[2]*r[2] + x[3]*r[3];
  }
  leftb[idx] = sl; rightb[idx] = sr;
}

__global__ void gat_final(const float* __restrict__ h1, const float* __restrict__ w2,
                          const float* __restrict__ b2, const int* __restrict__ batch,
                          float* __restrict__ out) {
  int n = blockIdx.x * blockDim.x + threadIdx.x;
  if (n >= kN) return;
  float s = b2[0];
#pragma unroll
  for (int k = 0; k < kH / 2; k += 4) {
    v4f x = *(const v4f*)(h1 + (size_t)n * (kH / 2) + k);
    v4f wv = *(const v4f*)(w2 + k);
    s += x[0]*wv[0] + x[1]*wv[1] + x[2]*wv[2] + x[3]*wv[3];
  }
  atomicAdd(&out[batch[n]], s);
}

// ---------------------------------------------------------------------------
extern "C" void kernel_launch(void* const* d_in, const int* in_sizes, int n_in,
                              void* d_out, int out_size, void* d_ws, size_t ws_size,
                              hipStream_t stream) {
  (void)in_sizes; (void)n_in; (void)out_size; (void)ws_size;
  const int*   z         = (const int*)d_in[0];
  const float* pos       = (const float*)d_in[1];
  const int*   batch     = (const int*)d_in[2];
  const int*   ei        = (const int*)d_in[3];
  const float* emb_table = (const float*)d_in[4];
  const float* lin_w     = (const float*)d_in[5];
  const float* attn_l    = (const float*)d_in[6];
  const float* attn_r    = (const float*)d_in[7];
  const float* attn_edge = (const float*)d_in[8];
  const float* mlp_w0    = (const float*)d_in[9];
  const float* mlp_b0    = (const float*)d_in[10];
  const float* mlp_w2    = (const float*)d_in[11];
  const float* mlp_b2    = (const float*)d_in[12];
  const float* v_lin1_w  = (const float*)d_in[13];
  const float* v_lin1_b  = (const float*)d_in[14];
  const float* v_lin2_w  = (const float*)d_in[15];
  const float* v_lin2_b  = (const float*)d_in[16];
  const float* u_lin1_w  = (const float*)d_in[17];
  const float* u_lin1_b  = (const float*)d_in[18];
  const float* u_lin2_w  = (const float*)d_in[19];
  const float* u_lin2_b  = (const float*)d_in[20];

  // ---- f32 scratch ----
  float* ws = (float*)d_ws;
  size_t o = 0;
  float* v    = ws + o; o += (size_t)kN * kH;      //  4.1 MB
  float* vl   = ws + o; o += (size_t)kN * kHF;     // 41.0 MB
  float* lb   = ws + o; o += (size_t)kN * kHD;
  float* rb   = ws + o; o += (size_t)kN * kHD;
  float* agg  = ws + o; o += (size_t)kN * kHF;     // 41.0 MB (L2-resident)
  float* tmp  = ws + o; o += (size_t)kN * kH;
  float* h1   = ws + o; o += (size_t)kN * (kH / 2);

  // ---- f16 scratch (after f32 region) ----
  _Float16* hws = (_Float16*)(ws + o);
  size_t ho = 0;
  _Float16* demb16 = hws + ho; ho += (size_t)kE * kGP;        // 8.2 MB
  _Float16* linT   = hws + ho; ho += (size_t)kL * kHF * kH;   // BT [1280,128] x2
  _Float16* w0T    = hws + ho; ho += (size_t)kL * kH * kGP;   // BT [128,64]   x2
  _Float16* w2T    = hws + ho; ho += (size_t)kL * kHF * kH;   // BT [1280,128] x2
  _Float16* v1T    = hws + ho; ho += (size_t)kL * kH * kHF;   // BT [128,1280] x2
  _Float16* v2T    = hws + ho; ho += (size_t)kL * kH * kH;    // BT [128,128]  x2
  _Float16* u1T    = hws + ho; ho += (size_t)(kH / 2) * kH;   // BT [64,128]

  const int* row = ei;
  const int* col = ei + kE;

  // ---- one-time prep: smearing, embedding gather, weight transposes ----
  gat_gsmear<<<(kE + 255) / 256, 256, 0, stream>>>(pos, row, col, demb16);
  gat_embed<<<(kN * kH + 255) / 256, 256, 0, stream>>>(z, emb_table, v);
  for (int l = 0; l < kL; ++l) {
    gat_conv_bt16<<<(kHF * kH + 255) / 256, 256, 0, stream>>>(
        lin_w + (size_t)l * kH * kHF, linT + (size_t)l * kHF * kH, kH, kHF, kH);
    gat_conv_bt16<<<(kH * kGP + 255) / 256, 256, 0, stream>>>(
        mlp_w0 + (size_t)l * kG * kF, w0T + (size_t)l * kH * kGP, kG, kF, kGP);
    gat_conv_bt16<<<(kHF * kH + 255) / 256, 256, 0, stream>>>(
        mlp_w2 + (size_t)l * kF * kHF, w2T + (size_t)l * kHF * kH, kF, kHF, kH);
    gat_conv_bt16<<<(kH * kHF + 255) / 256, 256, 0, stream>>>(
        v_lin1_w + (size_t)l * kHF * kH, v1T + (size_t)l * kH * kHF, kHF, kH, kHF);
    gat_conv_bt16<<<(kH * kH + 255) / 256, 256, 0, stream>>>(
        v_lin2_w + (size_t)l * kH * kH, v2T + (size_t)l * kH * kH, kH, kH, kH);
  }
  gat_conv_bt16<<<((kH / 2) * kH + 255) / 256, 256, 0, stream>>>(
      u_lin1_w, u1T, kH, kH / 2, kH);

  for (int l = 0; l < kL; ++l) {
    // vl = v @ lin_w[l]   [8000,128]x[128,1280]
    {
      int waves = (kN / 16) * (kHF / 64);
      gat_gemm_wmma<kH, 0, false, false><<<(waves + 7) / 8, 256, 0, stream>>>(
          v, linT + (size_t)l * kHF * kH, nullptr, nullptr, vl, kN, kHF);
    }
    gat_leftright<<<(kN * kHD + 255) / 256, 256, 0, stream>>>(
        vl, attn_l + (size_t)l * kHD * kF, attn_r + (size_t)l * kHD * kF, lb, rb);

    hipMemsetAsync(agg, 0, (size_t)kN * kHF * sizeof(float), stream);

    gat_edge_fused<<<kE / 16, 256, 0, stream>>>(
        demb16, row, col,
        w0T + (size_t)l * kH * kGP, mlp_b0 + (size_t)l * kF,
        w2T + (size_t)l * kHF * kH, mlp_b2 + (size_t)l * kHF,
        attn_edge + (size_t)l * kHD * kF, lb, rb, vl, agg);

    // tmp = ssp(agg @ v_lin1_w + b1) ; v += tmp @ v_lin2_w + b2
    {
      int waves = (kN / 16) * (kH / 64);
      gat_gemm_wmma<kHF, 1, true, false><<<(waves + 7) / 8, 256, 0, stream>>>(
          agg, v1T + (size_t)l * kH * kHF, v_lin1_b + (size_t)l * kH,
          nullptr, tmp, kN, kH);
      gat_gemm_wmma<kH, 0, true, true><<<(waves + 7) / 8, 256, 0, stream>>>(
          tmp, v2T + (size_t)l * kH * kH, v_lin2_b + (size_t)l * kH,
          v, v, kN, kH);
    }
  }

  // h1 = ssp(v @ u_lin1_w + b)   [8000,128]x[128,64]
  {
    int waves = (kN / 16) * ((kH / 2) / 64);
    gat_gemm_wmma<kH, 1, true, false><<<(waves + 7) / 8, 256, 0, stream>>>(
        v, u1T, u_lin1_b, nullptr, h1, kN, kH / 2);
  }

  hipMemsetAsync(d_out, 0, sizeof(float) * kNG, stream);
  gat_final<<<(kN + 255) / 256, 256, 0, stream>>>(h1, u_lin2_w, u_lin2_b, batch,
                                                  (float*)d_out);
}
